// DGCNN_Transformer_65970697666719
// MI455X (gfx1250) — matrix operand order repaired and checked by reading, not compiled, not run on hardware
//
#include <hip/hip_runtime.h>
#include <math.h>

// ---------------------------------------------------------------------------
// DGCNN / Point-Transformer forward for MI455X (gfx1250).
//  * All 1x1 convs -> bf16 WMMA GEMMs (v_wmma_f32_16x16x32_bf16, f32 accum),
//    register-blocked 2x2 (32x32 output per wave) so each A/B fragment feeds
//    two WMMAs -> ~2x arithmetic intensity vs 1 tile/wave.
//  * Edge-conv decomposed by linearity: kf(n,k) = KD[j]-KD[n]+kc[n] where
//    KD = wk_left@h + wp@pts (positional GEMM accumulated into KD).
//  * KNN: per-point LDS distance row + 20 parallel argmin passes.
//  * Grouped attention: wave32 shuffle reductions (d=8/16/32 lanes),
//    prefetch of next gathered columns (global_prefetch_b8).
// Workspace use ~130MB, all hot intermediates fit in the 192MB L2.
// ---------------------------------------------------------------------------

#define BATCH 4
#define NPTS  4096
#define KNN_K 20
#define NGROUPS 8
#define BN_EPS 1e-5f

typedef __attribute__((ext_vector_type(16))) __bf16 v16bf;
typedef __attribute__((ext_vector_type(8)))  float  v8f;

struct Bits32 { uint4 lo, hi; };   // 32 bytes = 16 bf16 lane-elements

__device__ __forceinline__ unsigned short f32_to_bf16(float f) {
    unsigned int u = __builtin_bit_cast(unsigned int, f);
    unsigned int rounding = 0x7FFFu + ((u >> 16) & 1u);
    return (unsigned short)((u + rounding) >> 16);
}

__device__ __forceinline__ float leaky(float v) { return v >= 0.f ? v : 0.2f * v; }

// --------------------------- data movement ---------------------------------

// replicate x (B,3,512) -> pts (B,3,4096)
__global__ void tile_kernel(const float* __restrict__ x, float* __restrict__ pts, int total) {
    int i = blockIdx.x * blockDim.x + threadIdx.x;
    if (i >= total) return;
    int n  = i & (NPTS - 1);
    int bc = i / NPTS;
    pts[i] = x[bc * 512 + (n & 511)];
}

// (B,C,N) f32 -> (B,N,Ck) bf16, zero-padded columns C..Ck
__global__ void cvt_act(const float* __restrict__ h, unsigned short* __restrict__ out,
                        int C, int Ck, int total) {
    int i = blockIdx.x * blockDim.x + threadIdx.x;
    if (i >= total) return;
    int kc = i % Ck;
    int rest = i / Ck;
    int n = rest % NPTS;
    int b = rest / NPTS;
    float v = (kc < C) ? h[((size_t)b * C + kc) * NPTS + n] : 0.f;
    out[i] = f32_to_bf16(v);
}

// W (O,totcols) f32, slice [coloff, coloff+cols) -> (O,Ck) bf16 zero-padded
__global__ void cvt_w(const float* __restrict__ w, unsigned short* __restrict__ out,
                      int O, int totcols, int coloff, int cols, int Ck) {
    int i = blockIdx.x * blockDim.x + threadIdx.x;
    if (i >= O * Ck) return;
    int k = i % Ck, o = i / Ck;
    float v = (k < cols) ? w[o * totcols + coloff + k] : 0.f;
    out[i] = f32_to_bf16(v);
}

// --------------------------- WMMA GEMM -------------------------------------
// C[b,O,N] (+)= Wb(O,Ck) @ Ht(b,N,Ck)^T.
// One wave computes a 32x32 output block = 2x2 WMMA tiles; each A/B fragment
// is consumed by two v_wmma_f32_16x16x32_bf16, halving fragment traffic.
// Lane layouts follow CDNA5 ISA §7.12.2 (16-bit A 16x32, B 32x16,
// f32 C/D 16x16).
__global__ __launch_bounds__(32)
void wmma_gemm_bf16(const unsigned short* __restrict__ Wb,
                    const unsigned short* __restrict__ Htb,
                    float* __restrict__ Cout,
                    int O, int Ck, int accumulate) {
    const int l    = threadIdx.x;
    const int nx   = blockIdx.x;    // 32-wide N block
    const int oy   = blockIdx.y;    // 32-wide O block
    const int b    = blockIdx.z;
    const int hi   = l >> 4;        // lane half
    const int lo16 = l & 15;
    const int m0   = oy * 32 + lo16;   // A row (first O tile) for this lane
    const int n0   = nx * 32 + lo16;   // B column (first N tile) for this lane

    // A lane element e -> k = (e&7) + 8*hi + 16*(e>>3)
    const unsigned short* arow0 = Wb + (size_t)m0 * Ck + (hi << 3);
    const unsigned short* arow1 = arow0 + (size_t)16 * Ck;          // O tile +16
    // B lane element e -> k = (e&7) + 8*hi + 16*(e>>3)
    const unsigned short* brow0 = Htb + ((size_t)b * NPTS + n0) * Ck + (hi << 3);
    const unsigned short* brow1 = brow0 + (size_t)16 * Ck;          // N tile +16

    v8f acc00 = {0.f, 0.f, 0.f, 0.f, 0.f, 0.f, 0.f, 0.f};
    v8f acc01 = acc00, acc10 = acc00, acc11 = acc00;
    if (accumulate) {
#pragma unroll
        for (int r = 0; r < 8; ++r) {
            const int mrow = r + (hi << 3);
            acc00[r] = Cout[((size_t)b * O + (oy * 32 + mrow)) * NPTS + n0];
            acc01[r] = Cout[((size_t)b * O + (oy * 32 + mrow)) * NPTS + n0 + 16];
            acc10[r] = Cout[((size_t)b * O + (oy * 32 + 16 + mrow)) * NPTS + n0];
            acc11[r] = Cout[((size_t)b * O + (oy * 32 + 16 + mrow)) * NPTS + n0 + 16];
        }
    }

    for (int kk = 0; kk < Ck; kk += 32) {
        Bits32 a0b, a1b, b0b, b1b;
        a0b.lo = *reinterpret_cast<const uint4*>(arow0 + kk);
        a0b.hi = *reinterpret_cast<const uint4*>(arow0 + kk + 16);
        a1b.lo = *reinterpret_cast<const uint4*>(arow1 + kk);
        a1b.hi = *reinterpret_cast<const uint4*>(arow1 + kk + 16);
        b0b.lo = *reinterpret_cast<const uint4*>(brow0 + kk);
        b0b.hi = *reinterpret_cast<const uint4*>(brow0 + kk + 16);
        b1b.lo = *reinterpret_cast<const uint4*>(brow1 + kk);
        b1b.hi = *reinterpret_cast<const uint4*>(brow1 + kk + 16);
        v16bf a0 = __builtin_bit_cast(v16bf, a0b);
        v16bf a1 = __builtin_bit_cast(v16bf, a1b);
        v16bf bm0 = __builtin_bit_cast(v16bf, b0b);
        v16bf bm1 = __builtin_bit_cast(v16bf, b1b);
        acc00 = __builtin_amdgcn_wmma_f32_16x16x32_bf16(
                    false, a0, false, bm0, (short)0, acc00, false, false);
        acc01 = __builtin_amdgcn_wmma_f32_16x16x32_bf16(
                    false, a0, false, bm1, (short)0, acc01, false, false);
        acc10 = __builtin_amdgcn_wmma_f32_16x16x32_bf16(
                    false, a1, false, bm0, (short)0, acc10, false, false);
        acc11 = __builtin_amdgcn_wmma_f32_16x16x32_bf16(
                    false, a1, false, bm1, (short)0, acc11, false, false);
    }

    // D layout: lane l, vgpr r -> M = r + 8*hi, N = lo16
#pragma unroll
    for (int r = 0; r < 8; ++r) {
        const int mrow = r + (hi << 3);
        Cout[((size_t)b * O + (oy * 32 + mrow)) * NPTS + n0]           = acc00[r];
        Cout[((size_t)b * O + (oy * 32 + mrow)) * NPTS + n0 + 16]      = acc01[r];
        Cout[((size_t)b * O + (oy * 32 + 16 + mrow)) * NPTS + n0]      = acc10[r];
        Cout[((size_t)b * O + (oy * 32 + 16 + mrow)) * NPTS + n0 + 16] = acc11[r];
    }
}

// --------------------------- KNN -------------------------------------------

__global__ void norm_kernel(const float* __restrict__ h, float* __restrict__ nrm,
                            int C, int total) {
    int i = blockIdx.x * blockDim.x + threadIdx.x;
    if (i >= total) return;
    int n = i % NPTS, b = i / NPTS;
    float s = 0.f;
    for (int c = 0; c < C; ++c) {
        float v = h[((size_t)b * C + c) * NPTS + n];
        s = __fmaf_rn(v, v, s);
    }
    nrm[i] = s;
}

// one workgroup per query point; distance row in LDS, 20 argmin passes
__global__ __launch_bounds__(256)
void knn_kernel(const float* __restrict__ h, const float* __restrict__ nrm,
                int* __restrict__ idxo, int C) {
    __shared__ float dist[NPTS];
    __shared__ float cf[128];
    __shared__ float rv[256];
    __shared__ int   ri[256];

    const int tid = threadIdx.x;
    const int pn  = blockIdx.x;
    const int b   = pn / NPTS;
    const int n   = pn % NPTS;

    for (int c = tid; c < C; c += 256) cf[c] = h[((size_t)b * C + c) * NPTS + n];
    __syncthreads();

    const float cn = nrm[pn];
    for (int j = tid; j < NPTS; j += 256) {
        float dot = 0.f;
        for (int c = 0; c < C; ++c)
            dot = __fmaf_rn(cf[c], h[((size_t)b * C + c) * NPTS + j], dot);
        float dd = cn - 2.f * dot + nrm[b * NPTS + j];
        if (j == n) dd += 1e9f;     // eye mask
        dist[j] = dd;
    }
    __syncthreads();

    for (int t = 0; t < KNN_K; ++t) {
        float best = 3e38f; int bi = NPTS;
        for (int j = tid; j < NPTS; j += 256) {
            float v = dist[j];
            if (v < best || (v == best && j < bi)) { best = v; bi = j; }
        }
        rv[tid] = best; ri[tid] = bi;
        __syncthreads();
        for (int s = 128; s > 0; s >>= 1) {
            if (tid < s) {
                float ov = rv[tid + s]; int oi = ri[tid + s];
                if (ov < rv[tid] || (ov == rv[tid] && oi < ri[tid])) {
                    rv[tid] = ov; ri[tid] = oi;
                }
            }
            __syncthreads();
        }
        if (tid == 0) {
            idxo[(size_t)pn * KNN_K + t] = ri[0];
            dist[ri[0]] = 3e38f;
        }
        __syncthreads();
    }
}

// --------------------------- attention -------------------------------------
// one workgroup per point, one thread per channel.
// kd/vd already contain the positional term (wp@pts accumulated by GEMM):
//   kf = kd[j] - kd[n] + kc[n],  vf = vd[j] - vd[n] + vc[n]
__global__ void attention_kernel(const float* __restrict__ q,
                                 const float* __restrict__ kd,
                                 const float* __restrict__ kc,
                                 const float* __restrict__ vd,
                                 const float* __restrict__ vc,
                                 const int* __restrict__ idx,
                                 float* __restrict__ y, int O) {
    __shared__ float vf[256 * KNN_K];
    __shared__ float lg[NGROUPS * KNN_K];
    __shared__ int   nbr[KNN_K];

    const int c  = threadIdx.x;
    const int pn = blockIdx.x;
    const int b  = pn / NPTS;
    const int n  = pn % NPTS;
    const int d  = O / NGROUPS;     // 8 / 8 / 16 / 32 -> lane-aligned groups
    const int g  = c / d;
    const float inv = rsqrtf((float)d);

    const size_t base = ((size_t)b * O + c) * NPTS + n;
    const float qc  = q[base];
    const float kdn = kd[base], kcn = kc[base];
    const float vdn = vd[base], vcn = vc[base];

    if (c < KNN_K) nbr[c] = idx[(size_t)pn * KNN_K + c];
    __syncthreads();

    const size_t rowbase = ((size_t)b * O + c) * NPTS;
    for (int k = 0; k < KNN_K; ++k) {
        const int j = nbr[k];
        if (k + 1 < KNN_K) {        // prefetch next gathered columns into L2/L0
            const size_t cjn = rowbase + nbr[k + 1];
            __builtin_prefetch(&kd[cjn], 0, 1);
            __builtin_prefetch(&vd[cjn], 0, 1);
        }
        const size_t cj = rowbase + j;
        float kf = kd[cj] - kdn + kcn;
        float vv = vd[cj] - vdn + vcn;
        vf[c * KNN_K + k] = vv;
        float p = qc * kf;
        for (int off = d >> 1; off > 0; off >>= 1)   // wave32 group reduction
            p += __shfl_xor(p, off, 32);
        if ((c % d) == 0) lg[g * KNN_K + k] = p * inv;
    }
    __syncthreads();

    if (c < NGROUPS) {              // softmax over K per group
        float mx = -3e38f;
        for (int k = 0; k < KNN_K; ++k) mx = fmaxf(mx, lg[c * KNN_K + k]);
        float s = 0.f;
        for (int k = 0; k < KNN_K; ++k) {
            float e = __expf(lg[c * KNN_K + k] - mx);
            lg[c * KNN_K + k] = e; s += e;
        }
        float rs = 1.f / s;
        for (int k = 0; k < KNN_K; ++k) lg[c * KNN_K + k] *= rs;
    }
    __syncthreads();

    float acc = 0.f;
    for (int k = 0; k < KNN_K; ++k)
        acc = __fmaf_rn(lg[g * KNN_K + k], vf[c * KNN_K + k], acc);
    y[base] = acc;
}

// --------------------------- batchnorm -------------------------------------

__global__ __launch_bounds__(256)
void bn_stats(const float* __restrict__ y, float* __restrict__ mean,
              float* __restrict__ var, int C) {
    __shared__ float ss[256], ss2[256];
    const int c = blockIdx.x, tid = threadIdx.x;
    const int M = BATCH * NPTS;
    float s = 0.f, s2 = 0.f;
    for (int i = tid; i < M; i += 256) {
        int b = i / NPTS, n = i % NPTS;
        float v = y[((size_t)b * C + c) * NPTS + n];
        s += v; s2 = __fmaf_rn(v, v, s2);
    }
    ss[tid] = s; ss2[tid] = s2;
    __syncthreads();
    for (int st = 128; st > 0; st >>= 1) {
        if (tid < st) { ss[tid] += ss[tid + st]; ss2[tid] += ss2[tid + st]; }
        __syncthreads();
    }
    if (tid == 0) {
        float m = ss[0] / (float)M;
        mean[c] = m;
        var[c]  = ss2[0] / (float)M - m * m;
    }
}

__global__ void bn_leaky(const float* __restrict__ y, float* __restrict__ h,
                         const float* __restrict__ mean, const float* __restrict__ var,
                         const float* __restrict__ g, const float* __restrict__ beta,
                         int C, int total) {
    int i = blockIdx.x * blockDim.x + threadIdx.x;
    if (i >= total) return;
    int c = (i / NPTS) % C;
    float v = (y[i] - mean[c]) * rsqrtf(var[c] + BN_EPS) * g[c] + beta[c];
    h[i] = leaky(v);
}

// BN + leaky + max over N  -> hmax (B, C)
__global__ __launch_bounds__(256)
void bn_leaky_max(const float* __restrict__ y, const float* __restrict__ mean,
                  const float* __restrict__ var, const float* __restrict__ g,
                  const float* __restrict__ beta, float* __restrict__ out, int C) {
    __shared__ float mx[256];
    const int c = blockIdx.x, b = blockIdx.y, tid = threadIdx.x;
    const float m = mean[c], rs = rsqrtf(var[c] + BN_EPS) * g[c], bb = beta[c];
    float best = -3e38f;
    for (int n = tid; n < NPTS; n += 256) {
        float v = (y[((size_t)b * C + c) * NPTS + n] - m) * rs + bb;
        best = fmaxf(best, leaky(v));
    }
    mx[tid] = best;
    __syncthreads();
    for (int st = 128; st > 0; st >>= 1) {
        if (tid < st) mx[tid] = fmaxf(mx[tid], mx[tid + st]);
        __syncthreads();
    }
    if (tid == 0) out[b * C + c] = mx[0];
}

// --------------------------- tail MLP --------------------------------------

__global__ void linear_k(const float* __restrict__ in, const float* __restrict__ w,
                         const float* __restrict__ bias, float* __restrict__ out,
                         int B_, int IN, int OUT) {
    int i = blockIdx.x * blockDim.x + threadIdx.x;
    if (i >= B_ * OUT) return;
    int b = i / OUT, o = i % OUT;
    float acc = bias ? bias[o] : 0.f;
    for (int c = 0; c < IN; ++c)
        acc = __fmaf_rn(in[b * IN + c], w[o * IN + c], acc);
    out[i] = acc;
}

__global__ void bn_batch_leaky(float* __restrict__ x, const float* __restrict__ g,
                               const float* __restrict__ beta, int B_, int F) {
    int o = blockIdx.x * blockDim.x + threadIdx.x;
    if (o >= F) return;
    float s = 0.f, s2 = 0.f;
    for (int b = 0; b < B_; ++b) { float v = x[b * F + o]; s += v; s2 += v * v; }
    float m = s / (float)B_;
    float vv = s2 / (float)B_ - m * m;
    float rs = rsqrtf(vv + BN_EPS) * g[o];
    for (int b = 0; b < B_; ++b) {
        float v = (x[b * F + o] - m) * rs + beta[o];
        x[b * F + o] = leaky(v);
    }
}

// --------------------------- host orchestration ----------------------------

extern "C" void kernel_launch(void* const* d_in, const int* in_sizes, int n_in,
                              void* d_out, int out_size, void* d_ws, size_t ws_size,
                              hipStream_t stream) {
    (void)in_sizes; (void)n_in; (void)out_size;
    const int B = BATCH, N = NPTS;

    const float* x = (const float*)d_in[0];
    struct Layer { const float *wq, *wk, *wv, *wp, *g, *b; int Cin, O; };
    Layer L[4];
    for (int i = 0; i < 4; ++i) {
        int base = 1 + i * 6;
        L[i].wq = (const float*)d_in[base + 0];
        L[i].wk = (const float*)d_in[base + 1];
        L[i].wv = (const float*)d_in[base + 2];
        L[i].wp = (const float*)d_in[base + 3];
        L[i].g  = (const float*)d_in[base + 4];
        L[i].b  = (const float*)d_in[base + 5];
    }
    L[0].Cin = 3;   L[0].O = 64;
    L[1].Cin = 64;  L[1].O = 64;
    L[2].Cin = 64;  L[2].O = 128;
    L[3].Cin = 128; L[3].O = 256;
    const float* w5     = (const float*)d_in[25];
    const float* g5     = (const float*)d_in[26];
    const float* b5     = (const float*)d_in[27];
    const float* lin1   = (const float*)d_in[28];
    const float* g6     = (const float*)d_in[29];
    const float* b6     = (const float*)d_in[30];
    const float* lin2   = (const float*)d_in[31];
    const float* lin2_b = (const float*)d_in[32];
    const float* g7     = (const float*)d_in[33];
    const float* b7     = (const float*)d_in[34];
    const float* lin3   = (const float*)d_in[35];
    const float* lin3_b = (const float*)d_in[36];

    // workspace bump allocator (256B aligned)
    char* wsb = (char*)d_ws;
    size_t off = 0;
    auto alloc = [&](size_t bytes) -> void* {
        off = (off + 255) & ~(size_t)255;
        void* p = wsb + off;
        off += bytes;
        return p;
    };
    const size_t fBN = (size_t)B * N;
    float*          pts  = (float*)alloc(fBN * 3 * 4);          // tiled points (B,3,N)
    unsigned short* ptb  = (unsigned short*)alloc(fBN * 32 * 2);// bf16 pts (B,N,32)
    float*          hbuf = (float*)alloc(fBN * 256 * 4);        // activations (B,<=256,N)
    float*          ybuf = (float*)alloc(fBN * 256 * 4);        // attention out
    float*          blob = (float*)alloc(fBN * 256 * 4 * 5);    // q,kd,kc,vd,vc | h5
    unsigned short* htb  = (unsigned short*)alloc(fBN * 256 * 2);
    unsigned short* wb   = (unsigned short*)alloc((size_t)1024 * 256 * 2);
    int*            idxb = (int*)alloc(fBN * KNN_K * 4);
    float*          nrm  = (float*)alloc(fBN * 4);
    float*          mean = (float*)alloc(1024 * 4);
    float*          var  = (float*)alloc(1024 * 4);
    float*          hmax = (float*)alloc((size_t)B * 1024 * 4);
    float*          fc1  = (float*)alloc((size_t)B * 512 * 4);
    float*          fc2  = (float*)alloc((size_t)B * 256 * 4);
    if (off > ws_size) return;   // workspace too small

    const size_t stride = fBN * 256;     // floats per q/k/v buffer
    float* qb  = blob + 0 * stride;
    float* kdb = blob + 1 * stride;
    float* kcb = blob + 2 * stride;
    float* vdb = blob + 3 * stride;
    float* vcb = blob + 4 * stride;

    auto gemm = [&](const unsigned short* W, const unsigned short* Hm, float* Co,
                    int O, int Ck, int accum) {
        dim3 grid(N / 32, O / 32, B);
        wmma_gemm_bf16<<<grid, dim3(32), 0, stream>>>(W, Hm, Co, O, Ck, accum);
    };
    auto cw = [&](const float* W, int O, int totc, int coloff, int cols, int Ck) {
        int tot = O * Ck;
        cvt_w<<<(tot + 255) / 256, 256, 0, stream>>>(W, wb, O, totc, coloff, cols, Ck);
    };

    // tile input, convert points to bf16 (padded to 32 cols)
    tile_kernel<<<(B * 3 * N + 255) / 256, 256, 0, stream>>>(x, pts, B * 3 * N);
    cvt_act<<<((int)(fBN * 32) + 255) / 256, 256, 0, stream>>>(pts, ptb, 3, 32, (int)(fBN * 32));

    const float* hcur = pts;   // layer-1 input features == points
    for (int i = 0; i < 4; ++i) {
        const int Cin = L[i].Cin, O = L[i].O;
        const int Ck  = (Cin < 32) ? 32 : Cin;

        // activations -> bf16 (B,N,Ck)
        cvt_act<<<((int)(fBN * Ck) + 255) / 256, 256, 0, stream>>>(
            hcur, htb, Cin, Ck, (int)(fBN * Ck));

        // KNN on current features
        norm_kernel<<<((int)fBN + 255) / 256, 256, 0, stream>>>(hcur, nrm, Cin, (int)fBN);
        knn_kernel<<<(int)fBN, 256, 0, stream>>>(hcur, nrm, idxb, Cin);

        // q = wq @ h
        cw(L[i].wq, O, Cin, 0, Cin, Ck);          gemm(wb, htb, qb,  O, Ck, 0);
        // kd = wk_left @ h  (+ wp @ pts)
        cw(L[i].wk, O, 2 * Cin, 0, Cin, Ck);      gemm(wb, htb, kdb, O, Ck, 0);
        cw(L[i].wp, O, 3, 0, 3, 32);              gemm(wb, ptb, kdb, O, 32, 1);
        // kc = wk_right @ h
        cw(L[i].wk, O, 2 * Cin, Cin, Cin, Ck);    gemm(wb, htb, kcb, O, Ck, 0);
        // vd = wv_left @ h (+ wp @ pts)
        cw(L[i].wv, O, 2 * Cin, 0, Cin, Ck);      gemm(wb, htb, vdb, O, Ck, 0);
        cw(L[i].wp, O, 3, 0, 3, 32);              gemm(wb, ptb, vdb, O, 32, 1);
        // vc = wv_right @ h
        cw(L[i].wv, O, 2 * Cin, Cin, Cin, Ck);    gemm(wb, htb, vcb, O, Ck, 0);

        attention_kernel<<<(int)fBN, O, 0, stream>>>(qb, kdb, kcb, vdb, vcb, idxb, ybuf, O);

        bn_stats<<<O, 256, 0, stream>>>(ybuf, mean, var, O);
        int tot = (int)(fBN * O);
        bn_leaky<<<(tot + 255) / 256, 256, 0, stream>>>(
            ybuf, hbuf, mean, var, L[i].g, L[i].b, O, tot);
        hcur = hbuf;
    }

    // h5 = w5 @ h   (1024 x 256 x 4096 GEMM), BN + leaky + max over N
    float* h5 = blob;   // 67MB, reuses q..vc region
    cvt_act<<<((int)(fBN * 256) + 255) / 256, 256, 0, stream>>>(
        hbuf, htb, 256, 256, (int)(fBN * 256));
    cw(w5, 1024, 256, 0, 256, 256);
    gemm(wb, htb, h5, 1024, 256, 0);
    bn_stats<<<1024, 256, 0, stream>>>(h5, mean, var, 1024);
    bn_leaky_max<<<dim3(1024, B), 256, 0, stream>>>(h5, mean, var, g5, b5, hmax, 1024);

    // MLP head
    linear_k<<<(B * 512 + 255) / 256, 256, 0, stream>>>(hmax, lin1, nullptr, fc1, B, 1024, 512);
    bn_batch_leaky<<<2, 256, 0, stream>>>(fc1, g6, b6, B, 512);
    linear_k<<<(B * 256 + 255) / 256, 256, 0, stream>>>(fc1, lin2, lin2_b, fc2, B, 512, 256);
    bn_batch_leaky<<<1, 256, 0, stream>>>(fc2, g7, b7, B, 256);
    linear_k<<<1, 256, 0, stream>>>(fc2, lin3, lin3_b, (float*)d_out, B, 256, 40);
}
